// DyGraphTransformer_87342454931929
// MI455X (gfx1250) — matrix operand order repaired and checked by reading, not compiled, not run on hardware
//
#include <hip/hip_runtime.h>
#include <math.h>

// Problem constants (match reference)
#define NN 4096
#define TT 1024
#define CC 3072
#define FF 128
#define HH 512
#define NHEADS 16
#define DD 32
#define NLAYERS 2
#define ET 65536
#define EC 98304
#define SCALE_F 0.17677669529663687f   // 32^-0.5

// GEMM tiling
#define MBLK 64
#define NBLK 128
#define KSTEP 32
#define LPAD 40   // LDS row stride in halfs (80B): n*20 mod 64 distinct -> conflict-free b128 reads

typedef __attribute__((ext_vector_type(16))) _Float16 v16h;
typedef __attribute__((ext_vector_type(8)))  _Float16 v8h;
typedef __attribute__((ext_vector_type(2)))  _Float16 v2h;
typedef __attribute__((ext_vector_type(8)))  float    v8f;
typedef __attribute__((ext_vector_type(4)))  float    v4f;

__device__ __forceinline__ float wave_reduce_add(float v) {
#pragma unroll
  for (int off = 16; off > 0; off >>= 1) v += __shfl_xor(v, off, 32);
  return v;
}

// ---------------------------------------------------------------------------
// LDS-staged WMMA GEMM: C[M x N] = act(A[M x K] @ B[K x N] + bias) (optionally +=)
// Block (256 thr / 8 waves) computes 64x128; wave computes 16x64 (4 accumulators).
// A staged row-major f16; B staged TRANSPOSED (n-major) f16 so fragments are
// contiguous ds_load_b128s. All B fragments preloaded -> back-to-back WMMA burst.
// Requires M%64==0, N%128==0, K%32==0 (true for all call sites here).
// ---------------------------------------------------------------------------
template <bool GELU, bool ACCUM>
__global__ void __launch_bounds__(256) gemm_wmma_kernel(
    const float* __restrict__ A, int lda,
    const float* __restrict__ B,
    const float* __restrict__ bias,
    float* __restrict__ C,
    int M, int N, int K) {
  __shared__ __align__(16) _Float16 Ah[MBLK][LPAD];
  __shared__ __align__(16) _Float16 Bh[NBLK][LPAD];

  const int tid  = threadIdx.x;
  const int lane = tid & 31;
  const int w    = tid >> 5;                 // wave 0..7
  const int nblk = N / NBLK;
  const int rowblk = (blockIdx.x / nblk) * MBLK;
  const int colblk = (blockIdx.x % nblk) * NBLK;

  const int wrow = (w >> 1) << 4;            // 0,16,32,48
  const int wcol = (w & 1) << 6;             // 0,64

  // staging coords
  const int a_row = tid >> 2;                // 0..63
  const int a_col = (tid & 3) << 3;          // 0,8,16,24
  const int b_k   = (tid & 15) << 1;         // 0,2,..,30
  const int b_n   = (tid >> 4) << 3;         // 0,8,..,120

  // fragment coords (CDNA5 ISA 7.12.2: 16-bit A 16x32 / B 32x16 / f32 C 16x16)
  const int fm = lane & 15;
  const int ka = (lane >> 4) << 3;           // +8 K for lanes 16..31 (A)
  const int kb = (lane >> 4) << 3;           // +8 K for lanes 16..31 (B)

  v8f acc[4] = {v8f{}, v8f{}, v8f{}, v8f{}};

  for (int k0 = 0; k0 < K; k0 += KSTEP) {
    // ---- stage A tile (64 x 32 f32 -> f16) ----
    {
      const float* pa = A + (size_t)(rowblk + a_row) * lda + k0 + a_col;
      const v4f f0 = *(const v4f*)pa;
      const v4f f1 = *(const v4f*)(pa + 4);
      v8h h;
#pragma unroll
      for (int j = 0; j < 4; ++j) { h[j] = (_Float16)f0[j]; h[4 + j] = (_Float16)f1[j]; }
      *(v8h*)&Ah[a_row][a_col] = h;
    }
    // ---- stage B tile (32 x 128 f32 -> f16, transposed to n-major) ----
    {
      const float* pb0 = B + (size_t)(k0 + b_k) * N + colblk + b_n;
      const float* pb1 = pb0 + N;
      const v4f r00 = *(const v4f*)pb0;
      const v4f r01 = *(const v4f*)(pb0 + 4);
      const v4f r10 = *(const v4f*)pb1;
      const v4f r11 = *(const v4f*)(pb1 + 4);
#pragma unroll
      for (int i = 0; i < 4; ++i) {
        v2h p; p[0] = (_Float16)r00[i]; p[1] = (_Float16)r10[i];
        *(v2h*)&Bh[b_n + i][b_k] = p;
      }
#pragma unroll
      for (int i = 0; i < 4; ++i) {
        v2h p; p[0] = (_Float16)r01[i]; p[1] = (_Float16)r11[i];
        *(v2h*)&Bh[b_n + 4 + i][b_k] = p;
      }
    }
    __syncthreads();

    // ---- preload all fragments (10 ds_load_b128 in flight), then WMMA burst ----
    const v8h alo = *(const v8h*)&Ah[wrow + fm][ka];        // K = ka..ka+7
    const v8h ahi = *(const v8h*)&Ah[wrow + fm][ka + 16];   // K = ka+16..ka+23
    v8h blo[4], bhi[4];
#pragma unroll
    for (int t = 0; t < 4; ++t) {
      blo[t] = *(const v8h*)&Bh[wcol + (t << 4) + fm][kb];      // K = kb..kb+7
      bhi[t] = *(const v8h*)&Bh[wcol + (t << 4) + fm][kb + 16]; // K = kb+16..kb+23
    }
    v16h a;
#pragma unroll
    for (int j = 0; j < 8; ++j) { a[j] = alo[j]; a[8 + j] = ahi[j]; }
#pragma unroll
    for (int t = 0; t < 4; ++t) {
      v16h b;
#pragma unroll
      for (int j = 0; j < 8; ++j) { b[j] = blo[t][j]; b[8 + j] = bhi[t][j]; }
      acc[t] = __builtin_amdgcn_wmma_f32_16x16x32_f16(
          false, a, false, b, (short)0, acc[t], false, false);
    }
    __syncthreads();
  }

  // ---- straight-line epilogue (templated: no branches) ----
  const int mbase = rowblk + wrow + ((lane >> 4) << 3);  // C VGPR i holds row mbase+i
  float* dst = C + (size_t)mbase * N + (colblk + wcol + fm);
#pragma unroll
  for (int t = 0; t < 4; ++t) {
    const float bv = bias[colblk + wcol + (t << 4) + fm];
#pragma unroll
    for (int i = 0; i < 8; ++i) {
      float v = acc[t][i] + bv;
      if (GELU) v = 0.5f * v * (1.0f + erff(v * 0.70710678118654752f)); // exact GELU
      float* p = dst + (size_t)i * N + (t << 4);
      if (ACCUM) *p += v; else *p = v;
    }
  }
}

// ---------------------------------------------------------------------------
// LayerNorm: one wave per row of H=512
// ---------------------------------------------------------------------------
__global__ void layernorm_kernel(const float* __restrict__ x,
                                 const float* __restrict__ g,
                                 const float* __restrict__ b,
                                 float* __restrict__ y, int rows) {
  const int wid  = (int)((blockIdx.x * blockDim.x + threadIdx.x) >> 5);
  const int lane = threadIdx.x & 31;
  if (wid >= rows) return;
  const float* xr = x + (size_t)wid * HH;
  float s = 0.0f;
#pragma unroll
  for (int i = lane; i < HH; i += 32) s += xr[i];
  const float mean = wave_reduce_add(s) * (1.0f / HH);
  float vs = 0.0f;
#pragma unroll
  for (int i = lane; i < HH; i += 32) { float d = xr[i] - mean; vs += d * d; }
  const float inv = rsqrtf(wave_reduce_add(vs) * (1.0f / HH) + 1e-5f);
  float* yr = y + (size_t)wid * HH;
#pragma unroll
  for (int i = lane; i < HH; i += 32) yr[i] = (xr[i] - mean) * inv * g[i] + b[i];
}

// ---------------------------------------------------------------------------
// Edge attention bias: one wave per edge.
// ---------------------------------------------------------------------------
__global__ void edge_bias_kernel(const int* __restrict__ rows, const int* __restrict__ cols,
                                 int row_off, int col_off,
                                 const int* __restrict__ edge_encodes,
                                 const int* __restrict__ edge_dist_encodes,
                                 const float* __restrict__ emb_tab,
                                 const float* __restrict__ dist_tab,
                                 const float* __restrict__ w_enc, const float* __restrict__ b_enc,
                                 const float* __restrict__ w_dist, const float* __restrict__ b_dist,
                                 float* __restrict__ bias_out, int E) {
  const int wid  = (int)((blockIdx.x * blockDim.x + threadIdx.x) >> 5);
  const int lane = threadIdx.x & 31;
  if (wid >= E) return;
  const int r = rows[wid] + row_off;
  const int c = cols[wid] + col_off;
  const size_t cell = (size_t)r * NN + c;
  const int i0 = edge_encodes[cell * 2 + 0];
  const int i1 = edge_encodes[cell * 2 + 1];
  const int i2 = edge_dist_encodes[cell];
  const float* t0 = emb_tab  + (size_t)i0 * HH;
  const float* t1 = emb_tab  + (size_t)i1 * HH;
  const float* t2 = dist_tab + (size_t)i2 * HH;

  float n0 = 0.0f, n1 = 0.0f, n2 = 0.0f;
#pragma unroll
  for (int i = lane; i < HH; i += 32) {
    float a = t0[i], b = t1[i], d = t2[i];
    n0 += a * a; n1 += b * b; n2 += d * d;
  }
  n0 = wave_reduce_add(n0); n1 = wave_reduce_add(n1); n2 = wave_reduce_add(n2);
  const float s0 = 0.5f / fmaxf(sqrtf(n0), 1.0f);   // 0.5 folds the WIN=2 mean
  const float s1 = 0.5f / fmaxf(sqrtf(n1), 1.0f);
  const float s2 = 1.0f / fmaxf(sqrtf(n2), 1.0f);

  float acc[NHEADS];
#pragma unroll
  for (int h = 0; h < NHEADS; ++h) acc[h] = 0.0f;
  for (int i = lane; i < HH; i += 32) {
    const float e1 = t0[i] * s0 + t1[i] * s1;
    const float e2 = t2[i] * s2;
    const float* wE = w_enc  + (size_t)i * NHEADS;
    const float* wD = w_dist + (size_t)i * NHEADS;
#pragma unroll
    for (int h = 0; h < NHEADS; ++h) acc[h] += e1 * wE[h] + e2 * wD[h];
  }
#pragma unroll
  for (int h = 0; h < NHEADS; ++h) acc[h] = wave_reduce_add(acc[h]);
  if (lane < NHEADS) {
    float outv = 0.0f;
#pragma unroll
    for (int h = 0; h < NHEADS; ++h) if (lane == h) outv = acc[h];
    bias_out[(size_t)wid * NHEADS + lane] = outv + b_enc[lane] + b_dist[lane];
  }
}

// ---------------------------------------------------------------------------
// Segment softmax, pass 1: scores + row-max (monotonic-int atomicMax)
// ---------------------------------------------------------------------------
__global__ void attn_score_kernel(const float* __restrict__ q, const float* __restrict__ k,
                                  const float* __restrict__ bias,
                                  const int* __restrict__ rows, const int* __restrict__ cols,
                                  float* __restrict__ s, int* __restrict__ mbuf, int E) {
  const int wid  = (int)((blockIdx.x * blockDim.x + threadIdx.x) >> 5);
  const int lane = threadIdx.x & 31;
  if (wid >= E) return;
  const int r = rows[wid], c = cols[wid];
  const float* qp = q + (size_t)r * HH;
  const float* kp = k + (size_t)c * HH;
#pragma unroll
  for (int h = 0; h < NHEADS; ++h) {
    float p = qp[h * DD + lane] * kp[h * DD + lane];
    p = wave_reduce_add(p);
    if (lane == 0) {
      const float sc = p * SCALE_F + bias[(size_t)wid * NHEADS + h];
      s[(size_t)wid * NHEADS + h] = sc;
      int si = __float_as_int(sc);
      si = (si >= 0) ? si : (si ^ 0x7FFFFFFF);      // monotonic float->int
      atomicMax(&mbuf[(size_t)r * NHEADS + h], si);
    }
  }
}

// pass 2: e = exp(s - m), sums via atomicAdd
__global__ void attn_exp_kernel(float* __restrict__ s, const int* __restrict__ mbuf,
                                float* __restrict__ sums, const int* __restrict__ rows,
                                int Eh) {
  const int i = (int)(blockIdx.x * blockDim.x + threadIdx.x);
  if (i >= Eh) return;
  const int e = i >> 4, h = i & 15;
  const int r = rows[e];
  int mi = mbuf[(size_t)r * NHEADS + h];
  mi = (mi >= 0) ? mi : (mi ^ 0x7FFFFFFF);
  const float ev = expf(s[i] - __int_as_float(mi));
  s[i] = ev;
  atomicAdd(&sums[(size_t)r * NHEADS + h], ev);
}

// pass 3: o[row] += (e / sum) * v[col]
__global__ void attn_scatter_kernel(const float* __restrict__ s, const float* __restrict__ sums,
                                    const float* __restrict__ v,
                                    const int* __restrict__ rows, const int* __restrict__ cols,
                                    float* __restrict__ o, int E) {
  const int wid  = (int)((blockIdx.x * blockDim.x + threadIdx.x) >> 5);
  const int lane = threadIdx.x & 31;
  if (wid >= E) return;
  const int r = rows[wid], c = cols[wid];
  const float* vp = v + (size_t)c * HH;
  float* op = o + (size_t)r * HH;
#pragma unroll
  for (int h = 0; h < NHEADS; ++h) {
    const float a = s[(size_t)wid * NHEADS + h] /
                    (sums[(size_t)r * NHEADS + h] + 1e-16f);
    atomicAdd(&op[h * DD + lane], a * vp[h * DD + lane]);
  }
}

__global__ void fill_u32_kernel(unsigned int* __restrict__ p, unsigned int v, long long n) {
  const long long i = blockIdx.x * (long long)blockDim.x + threadIdx.x;
  if (i < n) p[i] = v;
}

// ---------------------------------------------------------------------------
// Host-side launch helpers
// ---------------------------------------------------------------------------
static inline unsigned wave_grid(long long waves) { return (unsigned)((waves + 7) / 8); }

static void launch_gemm(const float* A, int lda, const float* B, const float* bias,
                        float* C, int M, int N, int K, int act, int accum,
                        hipStream_t stream) {
  const dim3 grid((unsigned)((M / MBLK) * (N / NBLK))), blk(256);
  if (act)
    gemm_wmma_kernel<true, false><<<grid, blk, 0, stream>>>(A, lda, B, bias, C, M, N, K);
  else if (accum)
    gemm_wmma_kernel<false, true><<<grid, blk, 0, stream>>>(A, lda, B, bias, C, M, N, K);
  else
    gemm_wmma_kernel<false, false><<<grid, blk, 0, stream>>>(A, lda, B, bias, C, M, N, K);
}

static void launch_fill(void* p, unsigned v, long long n, hipStream_t stream) {
  fill_u32_kernel<<<dim3((unsigned)((n + 255) / 256)), dim3(256), 0, stream>>>(
      (unsigned int*)p, v, n);
}

extern "C" void kernel_launch(void* const* d_in, const int* in_sizes, int n_in,
                              void* d_out, int out_size, void* d_ws, size_t ws_size,
                              hipStream_t stream) {
  const float* node_feats = (const float*)d_in[0];
  const float* emb_tab    = (const float*)d_in[1];
  const float* dist_tab   = (const float*)d_in[2];
  const float* w_node     = (const float*)d_in[3];
  const float* b_node     = (const float*)d_in[4];
  const float* w_enc      = (const float*)d_in[5];
  const float* b_enc      = (const float*)d_in[6];
  const float* w_dist     = (const float*)d_in[7];
  const float* b_dist     = (const float*)d_in[8];
  const float* ln1_g      = (const float*)d_in[9];
  const float* ln1_b      = (const float*)d_in[10];
  const float* attn1_w    = (const float*)d_in[11];
  const float* attn1_b    = (const float*)d_in[12];
  const float* attn2_w    = (const float*)d_in[13];
  const float* attn2_b    = (const float*)d_in[14];
  const float* ln2_g      = (const float*)d_in[15];
  const float* ln2_b      = (const float*)d_in[16];
  const float* ffn_w1     = (const float*)d_in[17];
  const float* ffn_b1     = (const float*)d_in[18];
  const float* ffn_w2     = (const float*)d_in[19];
  const float* ffn_b2     = (const float*)d_in[20];
  const int* edge_encodes      = (const int*)d_in[21];
  const int* edge_dist_encodes = (const int*)d_in[22];
  const int* t_row = (const int*)d_in[23];
  const int* t_col = (const int*)d_in[24];
  const int* c_row = (const int*)d_in[25];
  const int* c_col = (const int*)d_in[26];

  float* x  = (float*)d_out;                       // residual stream == final output
  float* ws = (float*)d_ws;
  size_t off = 0;
  float* y    = ws + off; off += (size_t)NN * HH;
  float* q1   = ws + off; off += (size_t)TT * HH;
  float* k1   = ws + off; off += (size_t)CC * HH;
  float* v1   = ws + off; off += (size_t)CC * HH;
  float* q2   = ws + off; off += (size_t)CC * HH;
  float* k2   = ws + off; off += (size_t)TT * HH;
  float* v2   = ws + off; off += (size_t)TT * HH;
  float* o1   = ws + off; off += (size_t)TT * HH;
  float* o2   = ws + off; off += (size_t)CC * HH;
  float* bias_t = ws + off; off += (size_t)ET * NHEADS;
  float* bias_c = ws + off; off += (size_t)EC * NHEADS;
  float* s1   = ws + off; off += (size_t)ET * NHEADS;
  float* s2   = ws + off; off += (size_t)EC * NHEADS;
  int*   m1   = (int*)(ws + off); off += (size_t)TT * NHEADS;
  float* sum1 = ws + off; off += (size_t)TT * NHEADS;
  int*   m2   = (int*)(ws + off); off += (size_t)CC * NHEADS;
  float* sum2 = ws + off; off += (size_t)CC * NHEADS;
  float* h1   = q1;   // FFN hidden aliases dead q1/k1/v1 region (N*H < T*H+2*C*H)

  const dim3 blk(256);

  // ---- edge attention biases (layer-invariant) ----
  edge_bias_kernel<<<dim3(wave_grid(ET)), blk, 0, stream>>>(
      t_row, t_col, 0, TT, edge_encodes, edge_dist_encodes,
      emb_tab, dist_tab, w_enc, b_enc, w_dist, b_dist, bias_t, ET);
  edge_bias_kernel<<<dim3(wave_grid(EC)), blk, 0, stream>>>(
      c_row, c_col, TT, 0, edge_encodes, edge_dist_encodes,
      emb_tab, dist_tab, w_enc, b_enc, w_dist, b_dist, bias_c, EC);

  // ---- input projection: x = node_feats @ w_node + b_node ----
  launch_gemm(node_feats, FF, w_node, b_node, x, NN, HH, FF, 0, 0, stream);

  for (int l = 0; l < NLAYERS; ++l) {
    const float* a1w = attn1_w + (size_t)l * 4 * HH * HH;
    const float* a1b = attn1_b + (size_t)l * 4 * HH;
    const float* a2w = attn2_w + (size_t)l * 4 * HH * HH;
    const float* a2b = attn2_b + (size_t)l * 4 * HH;

    // y = LN1(x)
    layernorm_kernel<<<dim3(wave_grid(NN)), blk, 0, stream>>>(
        x, ln1_g + (size_t)l * HH, ln1_b + (size_t)l * HH, y, NN);

    const float* y_t = y;
    const float* y_c = y + (size_t)TT * HH;

    // q/k/v projections (all WMMA)
    launch_gemm(y_t, HH, a1w + 0 * (size_t)HH * HH, a1b + 0 * HH, q1, TT, HH, HH, 0, 0, stream);
    launch_gemm(y_c, HH, a1w + 1 * (size_t)HH * HH, a1b + 1 * HH, k1, CC, HH, HH, 0, 0, stream);
    launch_gemm(y_c, HH, a1w + 2 * (size_t)HH * HH, a1b + 2 * HH, v1, CC, HH, HH, 0, 0, stream);
    launch_gemm(y_c, HH, a2w + 0 * (size_t)HH * HH, a2b + 0 * HH, q2, CC, HH, HH, 0, 0, stream);
    launch_gemm(y_t, HH, a2w + 1 * (size_t)HH * HH, a2b + 1 * HH, k2, TT, HH, HH, 0, 0, stream);
    launch_gemm(y_t, HH, a2w + 2 * (size_t)HH * HH, a2b + 2 * HH, v2, TT, HH, HH, 0, 0, stream);

    // reset segment stats / outputs
    launch_fill(m1, 0x80000000u, (long long)TT * NHEADS, stream);
    launch_fill(m2, 0x80000000u, (long long)CC * NHEADS, stream);
    launch_fill(sum1, 0u, (long long)TT * NHEADS, stream);
    launch_fill(sum2, 0u, (long long)CC * NHEADS, stream);
    launch_fill(o1, 0u, (long long)TT * HH, stream);
    launch_fill(o2, 0u, (long long)CC * HH, stream);

    // sparse attention, target side (q over T rows, k/v over C rows)
    attn_score_kernel<<<dim3(wave_grid(ET)), blk, 0, stream>>>(q1, k1, bias_t, t_row, t_col, s1, m1, ET);
    attn_exp_kernel<<<dim3((ET * NHEADS + 255) / 256), blk, 0, stream>>>(s1, m1, sum1, t_row, ET * NHEADS);
    attn_scatter_kernel<<<dim3(wave_grid(ET)), blk, 0, stream>>>(s1, sum1, v1, t_row, t_col, o1, ET);

    // sparse attention, context side (q over C rows, k/v over T rows)
    attn_score_kernel<<<dim3(wave_grid(EC)), blk, 0, stream>>>(q2, k2, bias_c, c_row, c_col, s2, m2, EC);
    attn_exp_kernel<<<dim3((EC * NHEADS + 255) / 256), blk, 0, stream>>>(s2, m2, sum2, c_row, EC * NHEADS);
    attn_scatter_kernel<<<dim3(wave_grid(EC)), blk, 0, stream>>>(s2, sum2, v2, c_row, c_col, o2, EC);

    // x = x + concat([out2 @ W2o, out1 @ W1o])  (out2 -> rows [0,C), out1 -> rows [C,N))
    launch_gemm(o2, HH, a2w + 3 * (size_t)HH * HH, a2b + 3 * HH, x,                   CC, HH, HH, 0, 1, stream);
    launch_gemm(o1, HH, a1w + 3 * (size_t)HH * HH, a1b + 3 * HH, x + (size_t)CC * HH, TT, HH, HH, 0, 1, stream);

    // FFN with exact GELU
    layernorm_kernel<<<dim3(wave_grid(NN)), blk, 0, stream>>>(
        x, ln2_g + (size_t)l * HH, ln2_b + (size_t)l * HH, y, NN);
    launch_gemm(y,  HH, ffn_w1 + (size_t)l * HH * HH, ffn_b1 + (size_t)l * HH, h1, NN, HH, HH, 1, 0, stream);
    launch_gemm(h1, HH, ffn_w2 + (size_t)l * HH * HH, ffn_b2 + (size_t)l * HH, x,  NN, HH, HH, 0, 1, stream);
  }

  (void)in_sizes; (void)n_in; (void)out_size; (void)ws_size;
}
